// ViTLayer_81209241633055
// MI455X (gfx1250) — matrix operand order repaired and checked by reading, not compiled, not run on hardware
//
#include <hip/hip_runtime.h>
#include <hip/hip_bf16.h>

typedef __attribute__((ext_vector_type(16))) _Float16 v16h;
typedef __attribute__((ext_vector_type(8)))  float    v8f;
typedef __attribute__((ext_vector_type(4)))  int      i4v;

#define WMMA_F32_F16(a, b, c) \
  __builtin_amdgcn_wmma_f32_16x16x32_f16(false, (a), false, (b), (short)0, (c), false, false)

#define AS1 __attribute__((address_space(1)))
#define AS3 __attribute__((address_space(3)))

#if defined(__HIP_DEVICE_COMPILE__) && defined(__has_builtin)
#if __has_builtin(__builtin_amdgcn_global_load_async_to_lds_b128)
#define HAVE_ASYNC_LDS 1
#endif
#endif

__device__ inline void async_lds_b128(const void* gsrc, void* ldst) {
#if defined(HAVE_ASYNC_LDS)
  __builtin_amdgcn_global_load_async_to_lds_b128(
      (AS1 i4v*)gsrc, (AS3 i4v*)ldst, 0, 0);
#else
  *(uint4*)ldst = *(const uint4*)gsrc;
#endif
}

__device__ inline void wait_async_lds() {
#if defined(HAVE_ASYNC_LDS)
#if __has_builtin(__builtin_amdgcn_s_wait_asynccnt)
  __builtin_amdgcn_s_wait_asynccnt(0);
#else
  asm volatile("s_wait_asynccnt 0x0" ::: "memory");
#endif
#endif
}

// ---------------------------------------------------------------------------
// Fragment loaders (wave32 layouts per CDNA5 ISA 7.12.2).
// A (16x32 f16): lanes 0-15 row M=lane, elems 0..7 -> K=0..7, 8..15 -> K=16..23
//                lanes 16-31 same rows,  elems 0..7 -> K=8..15, 8..15 -> K=24..31
__device__ inline v16h load_frag_a(const _Float16* As, int lda, int lane) {
  int row   = lane & 15;
  int kbase = (lane >> 4) * 8;
  const _Float16* p = As + row * lda + kbase;
  union { uint4 u[2]; v16h h; } u;
  u.u[0] = *(const uint4*)(p);        // K = kbase .. kbase+7
  u.u[1] = *(const uint4*)(p + 16);   // K = kbase+16 .. kbase+23
  return u.h;
}

// B (32x16 f16) from an LDS tile stored TRANSPOSED as Bt[n][k] (row stride ldb):
// lanes 0-15: N=lane, K=0..7,16..23 ; lanes 16-31: N=lane-16, K=8..15,24..31 (matches A).
__device__ inline v16h load_frag_bt(const _Float16* Bt, int ldb, int lane) {
  int n     = lane & 15;
  int kbase = (lane >> 4) * 8;
  const _Float16* p = Bt + n * ldb + kbase;
  union { uint4 u[2]; v16h h; } u;
  u.u[0] = *(const uint4*)(p);
  u.u[1] = *(const uint4*)(p + 16);
  return u.h;
}
// C/D (16x16 f32): reg j -> row = j + 8*(lane>=16), col = lane&15.

// ---------------------------------------------------------------------------
__global__ void cvt_f16_kernel(const float* __restrict__ s, _Float16* __restrict__ d, int n) {
  int i = (blockIdx.x * 256 + threadIdx.x) * 4;
  if (i < n) {
    float4 v = *(const float4*)(s + i);
    union { uint2 u; _Float16 h[4]; } o;
    o.h[0] = (_Float16)v.x; o.h[1] = (_Float16)v.y;
    o.h[2] = (_Float16)v.z; o.h[3] = (_Float16)v.w;
    *(uint2*)(d + i) = o.u;
  }
}

// ---------------------------------------------------------------------------
// LayerNorm over D=1024, one row per 256-thread block, f16 output.
__global__ __launch_bounds__(256) void ln_kernel(const float* __restrict__ x,
                                                 const float* __restrict__ g,
                                                 const float* __restrict__ beta,
                                                 _Float16* __restrict__ out) {
  __shared__ float red[256];
  int row = blockIdx.x, tid = threadIdx.x;
  const float* xr = x + (size_t)row * 1024;
  float4 v = ((const float4*)xr)[tid];
  red[tid] = (v.x + v.y) + (v.z + v.w);
  __syncthreads();
  for (int st = 128; st > 0; st >>= 1) { if (tid < st) red[tid] += red[tid + st]; __syncthreads(); }
  float mean = red[0] * (1.0f / 1024.0f);
  __syncthreads();
  float dx = v.x - mean, dy = v.y - mean, dz = v.z - mean, dw = v.w - mean;
  red[tid] = dx * dx + dy * dy + dz * dz + dw * dw;
  __syncthreads();
  for (int st = 128; st > 0; st >>= 1) { if (tid < st) red[tid] += red[tid + st]; __syncthreads(); }
  float rstd = rsqrtf(red[0] * (1.0f / 1024.0f) + 1e-5f);
  float4 gv = ((const float4*)g)[tid];
  float4 bv = ((const float4*)beta)[tid];
  union { uint2 u; _Float16 h[4]; } o;
  o.h[0] = (_Float16)(dx * rstd * gv.x + bv.x);
  o.h[1] = (_Float16)(dy * rstd * gv.y + bv.y);
  o.h[2] = (_Float16)(dz * rstd * gv.z + bv.z);
  o.h[3] = (_Float16)(dw * rstd * gv.w + bv.w);
  ((uint2*)(out + (size_t)row * 1024))[tid] = o.u;
}

// ---------------------------------------------------------------------------
// WMMA GEMM: C[M,N] = A16[M,K] @ W16[K,N] + bias.  BM=128 BN=64 BK=64, 8 waves.
// mode 0: head-scatter f16 -> out[((b*16+h)*1024+nseq)*64+s], h=n%16, s=n/16
// mode 1: silu f16 row-major
// mode 2: f32 out = val + res (residual), row-major
__global__ __launch_bounds__(256) void gemm_wmma_kernel(
    const _Float16* __restrict__ A, const _Float16* __restrict__ W,
    const float* __restrict__ bias, int M, int N, int K,
    int mode, void* __restrict__ out, const float* __restrict__ res) {
  __shared__ _Float16 As[128 * 64];   // [m][k]  16KB
  __shared__ _Float16 Bst[64 * 64];   // transposed: [n][k]  8KB
  int tid = threadIdx.x, lane = tid & 31, w = tid >> 5;
  int wm = w >> 1, wn = w & 1;                   // wave tile: 32x32
  int m0 = blockIdx.x * 128, n0 = blockIdx.y * 64;
  v8f acc[2][2] = {};

  for (int k0 = 0; k0 < K; k0 += 64) {
    {  // A: 128x64, each thread 32 halves (4x uint4)
      int r = tid >> 1, off = (tid & 1) * 32;
      const uint4* src = (const uint4*)(A + (size_t)(m0 + r) * K + k0 + off);
      uint4* dst = (uint4*)(As + r * 64 + off);
      dst[0] = src[0]; dst[1] = src[1]; dst[2] = src[2]; dst[3] = src[3];
    }
    {  // B: 64x64 read coalesced, stored transposed [n][k]
      int r = tid >> 3, off = (tid & 7) * 8;
#pragma unroll
      for (int rr = 0; rr < 2; ++rr) {
        int kk = r + rr * 32;
        union { uint4 u; _Float16 h[8]; } t;
        t.u = *(const uint4*)(W + (size_t)(k0 + kk) * N + n0 + off);
#pragma unroll
        for (int e = 0; e < 8; ++e) Bst[(off + e) * 64 + kk] = t.h[e];
      }
    }
    __syncthreads();
#pragma unroll
    for (int kk = 0; kk < 64; kk += 32) {
      v16h a0 = load_frag_a(As + (wm * 32 + 0)  * 64 + kk, 64, lane);
      v16h a1 = load_frag_a(As + (wm * 32 + 16) * 64 + kk, 64, lane);
      v16h b0 = load_frag_bt(Bst + (wn * 32 + 0)  * 64 + kk, 64, lane);
      v16h b1 = load_frag_bt(Bst + (wn * 32 + 16) * 64 + kk, 64, lane);
      acc[0][0] = WMMA_F32_F16(a0, b0, acc[0][0]);
      acc[0][1] = WMMA_F32_F16(a0, b1, acc[0][1]);
      acc[1][0] = WMMA_F32_F16(a1, b0, acc[1][0]);
      acc[1][1] = WMMA_F32_F16(a1, b1, acc[1][1]);
    }
    __syncthreads();
  }

  int half = lane >> 4, ln = lane & 15;
#pragma unroll
  for (int i = 0; i < 2; ++i)
#pragma unroll
    for (int j = 0; j < 2; ++j)
#pragma unroll
      for (int r = 0; r < 8; ++r) {
        int m = m0 + wm * 32 + i * 16 + half * 8 + r;
        int n = n0 + wn * 32 + j * 16 + ln;
        float val = acc[i][j][r] + bias[n];
        if (mode == 0) {
          int bimg = m >> 10, nseq = m & 1023;
          int h = n & 15, s = n >> 4;
          ((_Float16*)out)[(((size_t)bimg * 16 + h) * 1024 + nseq) * 64 + s] = (_Float16)val;
        } else if (mode == 1) {
          float sig = 1.0f / (1.0f + __expf(-val));
          ((_Float16*)out)[(size_t)m * N + n] = (_Float16)(val * sig);
        } else {
          size_t idx = (size_t)m * N + n;
          ((float*)out)[idx] = val + res[idx];
        }
      }
}

// ---------------------------------------------------------------------------
// Attention: one block per (bh, 32-row tile).  8 waves (2 row-tiles x 4 col-quads).
// LDS: K staged whole [1024 keys][64 hs] (128KB), later reused as Vt[64][1024];
//      Q tile [32][64]; P [32][1024] f16; reduction scratch.
#define ATTN_SMEM_BYTES 201472
__global__ __launch_bounds__(256) void attn_kernel(
    const _Float16* __restrict__ Qh, const _Float16* __restrict__ Kh,
    const _Float16* __restrict__ Vh, const float* __restrict__ x,
    float* __restrict__ x2) {
  extern __shared__ char smem[];
  _Float16* KV  = (_Float16*)smem;            // 65536 halves
  _Float16* Qs  = KV + 65536;                 // 2048 halves
  _Float16* Ps  = Qs + 2048;                  // 32768 halves
  float* redbuf = (float*)(Ps + 32768);       // [32 rows][4 quads]
  float* rowmax = redbuf + 128;               // [32]
  float* rowsum = rowmax + 32;                // [32]

  int tid = threadIdx.x, lane = tid & 31, w = tid >> 5;
  int half = lane >> 4, ln = lane & 15;
  int bh = blockIdx.y, row0 = blockIdx.x * 32;
  const _Float16* Qg = Qh + (size_t)bh * 1024 * 64;
  const _Float16* Kg = Kh + (size_t)bh * 1024 * 64;
  const _Float16* Vg = Vh + (size_t)bh * 1024 * 64;

  // stage Q tile (32x64) and all of K (1024x64, natural layout = Bt for QK^T)
  // via CDNA5 async global->LDS (ASYNCcnt) when available
  async_lds_b128((const char*)(Qg + (size_t)row0 * 64) + tid * 16,
                 (char*)Qs + tid * 16);
  for (int i = tid; i < 8192; i += 256)
    async_lds_b128((const char*)Kg + (size_t)i * 16, (char*)KV + (size_t)i * 16);
  wait_async_lds();
  __syncthreads();

  int rw = w >> 2, nq = w & 3;
  v16h qa0 = load_frag_a(Qs + (rw * 16) * 64 + 0,  64, lane);
  v16h qa1 = load_frag_a(Qs + (rw * 16) * 64 + 32, 64, lane);

  v8f S[16];
#pragma unroll
  for (int t = 0; t < 16; ++t) {
    int nt = nq * 16 + t;
    v8f a = {};
    v16h b0 = load_frag_bt(KV + (nt * 16) * 64 + 0,  64, lane);
    v16h b1 = load_frag_bt(KV + (nt * 16) * 64 + 32, 64, lane);
    a = WMMA_F32_F16(qa0, b0, a);
    a = WMMA_F32_F16(qa1, b1, a);
    S[t] = a;
  }

  // ---- softmax: row max ----
#pragma unroll
  for (int r = 0; r < 8; ++r) {
    float m = -1e30f;
#pragma unroll
    for (int t = 0; t < 16; ++t) m = fmaxf(m, S[t][r]);
    for (int off = 1; off < 16; off <<= 1) m = fmaxf(m, __shfl_xor(m, off, 32));
    if (ln == 0) redbuf[(rw * 16 + half * 8 + r) * 4 + nq] = m;
  }
  __syncthreads();
  if (tid < 32)
    rowmax[tid] = fmaxf(fmaxf(redbuf[tid * 4 + 0], redbuf[tid * 4 + 1]),
                        fmaxf(redbuf[tid * 4 + 2], redbuf[tid * 4 + 3]));
  __syncthreads();

  // ---- exp, row sum, write P(f16) ----
  const float scale = 0.125f;  // 1/sqrt(64)
#pragma unroll
  for (int r = 0; r < 8; ++r) {
    int row = rw * 16 + half * 8 + r;
    float m = rowmax[row];
    float s = 0.0f;
#pragma unroll
    for (int t = 0; t < 16; ++t) {
      float p = __expf((S[t][r] - m) * scale);
      s += p;
      Ps[row * 1024 + (nq * 16 + t) * 16 + ln] = (_Float16)p;
    }
    for (int off = 1; off < 16; off <<= 1) s += __shfl_xor(s, off, 32);
    if (ln == 0) redbuf[row * 4 + nq] = s;
  }
  __syncthreads();  // scores done -> safe to overwrite KV with V
  if (tid < 32)
    rowsum[tid] = (redbuf[tid * 4 + 0] + redbuf[tid * 4 + 1]) +
                  (redbuf[tid * 4 + 2] + redbuf[tid * 4 + 3]);

  // stage V transposed: Vt[hs][key] so P@V B-frags are contiguous
  for (int i = tid; i < 8192; i += 256) {
    union { uint4 u; _Float16 h[8]; } t;
    t.u = ((const uint4*)Vg)[i];
    int key = i >> 3, hs0 = (i & 7) * 8;
#pragma unroll
    for (int e = 0; e < 8; ++e) KV[(hs0 + e) * 1024 + key] = t.h[e];
  }
  __syncthreads();

  // ---- O = P @ V : 8 C-tiles (2 row-tiles x 4 hs-tiles), one per wave ----
  int nt2 = w & 3;
  v8f O = {};
  for (int ks = 0; ks < 1024; ks += 32) {
    v16h a = load_frag_a(Ps + (rw * 16) * 1024 + ks, 1024, lane);
    v16h b = load_frag_bt(KV + (nt2 * 16) * 1024 + ks, 1024, lane);
    O = WMMA_F32_F16(a, b, O);
  }

  int bimg = bh >> 4, hidx = bh & 15;
#pragma unroll
  for (int r = 0; r < 8; ++r) {
    int row = rw * 16 + half * 8 + r;
    int nseq = row0 + row;
    int hs = nt2 * 16 + ln;
    float o = O[r] / rowsum[row];
    size_t idx = ((size_t)bimg * 1024 + nseq) * 1024 + hidx * 64 + hs;
    x2[idx] = x[idx] + o;  // fused residual
  }
}

// ---------------------------------------------------------------------------
extern "C" void kernel_launch(void* const* d_in, const int* in_sizes, int n_in,
                              void* d_out, int out_size, void* d_ws, size_t ws_size,
                              hipStream_t stream) {
  const float* x     = (const float*)d_in[0];
  const float* ln1_g = (const float*)d_in[1];
  const float* ln1_b = (const float*)d_in[2];
  const float* ln2_g = (const float*)d_in[3];
  const float* ln2_b = (const float*)d_in[4];
  const float* wq = (const float*)d_in[5];  const float* bq = (const float*)d_in[6];
  const float* wk = (const float*)d_in[7];  const float* bk = (const float*)d_in[8];
  const float* wv = (const float*)d_in[9];  const float* bv = (const float*)d_in[10];
  const float* w1 = (const float*)d_in[11]; const float* b1 = (const float*)d_in[12];
  const float* w2 = (const float*)d_in[13]; const float* b2 = (const float*)d_in[14];

  char* ws = (char*)d_ws;
  // workspace layout (byte offsets)
  _Float16* wq16 = (_Float16*)(ws + 0);          // 1M halves
  _Float16* wk16 = (_Float16*)(ws + 2097152);
  _Float16* wv16 = (_Float16*)(ws + 4194304);
  _Float16* w116 = (_Float16*)(ws + 6291456);    // 4M halves
  _Float16* w216 = (_Float16*)(ws + 14680064);   // 4M halves
  _Float16* h16  = (_Float16*)(ws + 23068672);   // 16M halves (LN1 out, reused for LN2 out)
  float*    x2   = (float*)   (ws + 56623104);   // 16M floats (x + attn)
  _Float16* Qh   = (_Float16*)(ws + 123731968);  // 16M halves
  _Float16* Kh   = (_Float16*)(ws + 157286400);
  _Float16* Vh   = (_Float16*)(ws + 190840832);
  _Float16* y1   = (_Float16*)(ws + 123731968);  // 64M halves, reuses dead Q/K/V
  float* out = (float*)d_out;

  (void)hipFuncSetAttribute((const void*)attn_kernel,
                            hipFuncAttributeMaxDynamicSharedMemorySize, ATTN_SMEM_BYTES);

  // 1) weights -> f16
  cvt_f16_kernel<<<1024, 256, 0, stream>>>(wq, wq16, 1048576);
  cvt_f16_kernel<<<1024, 256, 0, stream>>>(wk, wk16, 1048576);
  cvt_f16_kernel<<<1024, 256, 0, stream>>>(wv, wv16, 1048576);
  cvt_f16_kernel<<<4096, 256, 0, stream>>>(w1, w116, 4194304);
  cvt_f16_kernel<<<4096, 256, 0, stream>>>(w2, w216, 4194304);

  // 2) LN1
  ln_kernel<<<16384, 256, 0, stream>>>(x, ln1_g, ln1_b, h16);

  // 3) QKV projections with head-scatter epilogue
  dim3 gqkv(128, 16);
  gemm_wmma_kernel<<<gqkv, 256, 0, stream>>>(h16, wq16, bq, 16384, 1024, 1024, 0, Qh, nullptr);
  gemm_wmma_kernel<<<gqkv, 256, 0, stream>>>(h16, wk16, bk, 16384, 1024, 1024, 0, Kh, nullptr);
  gemm_wmma_kernel<<<gqkv, 256, 0, stream>>>(h16, wv16, bv, 16384, 1024, 1024, 0, Vh, nullptr);

  // 4) attention + residual -> x2
  attn_kernel<<<dim3(32, 256), 256, ATTN_SMEM_BYTES, stream>>>(Qh, Kh, Vh, x, x2);

  // 5) LN2
  ln_kernel<<<16384, 256, 0, stream>>>(x2, ln2_g, ln2_b, h16);

  // 6) MLP fc1 + SiLU
  gemm_wmma_kernel<<<dim3(128, 64), 256, 0, stream>>>(h16, w116, b1, 16384, 4096, 1024, 1, y1, nullptr);

  // 7) MLP fc2 + residual -> out
  gemm_wmma_kernel<<<dim3(128, 16), 256, 0, stream>>>(y1, w216, b2, 16384, 1024, 4096, 2, out, x2);
}
